// MambaBlock_78709570667247
// MI455X (gfx1250) — matrix operand order repaired and checked by reading, compiled, not run on hardware
//
#include <hip/hip_runtime.h>
#include <math.h>

// ---------------- problem geometry ----------------
#define D_MODEL 128
#define BATCH   2
#define SEQ     1024
#define ROWS    (BATCH * SEQ)      // 2048 token rows
#define CHUNK   256                // seq chunk for expm/scan interleave
#define NCHUNK  (SEQ / CHUNK)      // 4
#define LDSS    132                // f32 LDS row stride (floats)
#define LDSB    136                // bf16 A-layout row stride (bf16 elems, even)
#define PSTR    132                // B-pair layout row stride (dwords), rows = K-pairs

typedef __attribute__((ext_vector_type(2)))  float    v2f;
typedef __attribute__((ext_vector_type(8)))  float    v8f;
typedef __attribute__((ext_vector_type(8)))  unsigned v8u;
typedef __attribute__((ext_vector_type(16))) __bf16   v16bf;

union FragCvt { v8u u; v16bf b; };

// ---------------- WMMA wrappers (CDNA5, wave32) ----------------
__device__ __forceinline__ v8f wmma_f32_4(v2f a, v2f b, v8f c) {
  return __builtin_amdgcn_wmma_f32_16x16x4_f32(false, a, false, b, (short)0, c, false, false);
}
__device__ __forceinline__ v8f wmma_bf16(v16bf a, v16bf b, v8f c) {
  return __builtin_amdgcn_wmma_f32_16x16x32_bf16(false, a, false, b, (short)0, c, false, false);
}

__device__ __forceinline__ float gelu_exact(float v) {
  return 0.5f * v * (1.0f + erff(v * 0.70710678118654752440f));
}

// =====================================================================
//  bf16x3 split-precision 128x128 matmul in LDS (8 waves / 256 thr)
//  D = alpha * (A x B) (+ I), A given as bf16 hi/lo in row-major A-layout,
//  B given as bf16 hi/lo in K-pair-packed B-layout ([64 kpairs][PSTR] dwords).
//  A-frag (16-bit 16x32): lane l: m=l&15, half=l>>4;
//    dword j   : kpair = kb*16 + half*8 + j
//  B-frag (32x16): dword j: kpair = kb*16 + half*8 + j, col = l&15
// =====================================================================
__device__ __forceinline__ void wg_matmul_bf16x3(const __bf16* __restrict__ AhA,
                                                 const __bf16* __restrict__ AlA,
                                                 const unsigned* __restrict__ BhP,
                                                 const unsigned* __restrict__ BlP,
                                                 float* __restrict__ Dm,
                                                 float alpha, bool addI) {
  const int lane = threadIdx.x & 31;
  const int wid  = threadIdx.x >> 5;      // 0..7 -> 16-row M tile
  const int half = lane >> 4;
  const int l15  = lane & 15;
  const int m    = wid * 16 + l15;

  v8f acc[8];
  #pragma unroll
  for (int nt = 0; nt < 8; ++nt) acc[nt] = (v8f){};

  const unsigned* rowH = (const unsigned*)(AhA + m * LDSB);
  const unsigned* rowL = (const unsigned*)(AlA + m * LDSB);

  #pragma unroll
  for (int kb = 0; kb < 4; ++kb) {        // K blocks of 32
    const int kd = kb * 16;               // dword (K-pair) base of the block
    FragCvt ah, al;
    #pragma unroll
    for (int j = 0; j < 4; ++j) {
      ah.u[j] = rowH[kd + half * 8 + j];
      al.u[j] = rowL[kd + half * 8 + j];
    }
    #pragma unroll
    for (int j = 0; j < 4; ++j) {
      ah.u[4 + j] = rowH[kd + half * 8 + 4 + j];
      al.u[4 + j] = rowL[kd + half * 8 + 4 + j];
    }
    const int kp0 = kd + half * 8;
    #pragma unroll
    for (int nt = 0; nt < 8; ++nt) {
      const int col = nt * 16 + l15;
      FragCvt bh, bl;
      #pragma unroll
      for (int j = 0; j < 8; ++j) {
        bh.u[j] = BhP[(kp0 + j) * PSTR + col];
        bl.u[j] = BlP[(kp0 + j) * PSTR + col];
      }
      acc[nt] = wmma_bf16(ah.b, bh.b, acc[nt]);  // hi*hi
      acc[nt] = wmma_bf16(ah.b, bl.b, acc[nt]);  // hi*lo
      acc[nt] = wmma_bf16(al.b, bh.b, acc[nt]);  // lo*hi
    }
  }
  #pragma unroll
  for (int nt = 0; nt < 8; ++nt) {
    const int col = nt * 16 + l15;
    #pragma unroll
    for (int v = 0; v < 8; ++v) {
      const int r = wid * 16 + v + half * 8;
      float val = acc[nt][v] * alpha;
      if (addI && r == col) val += 1.0f;
      Dm[r * LDSS + col] = val;
    }
  }
}

// split f32 -> bf16 hi/lo, write both A-layout and B-pair-layout copies
__device__ __forceinline__ void split_store(float x, int r, int c,
                                            __bf16* __restrict__ Ah,
                                            __bf16* __restrict__ Al,
                                            __bf16* __restrict__ Bh,
                                            __bf16* __restrict__ Bl) {
  const __bf16 h = (__bf16)x;
  const __bf16 l = (__bf16)(x - (float)h);
  Ah[r * LDSB + c] = h;
  Al[r * LDSB + c] = l;
  const int kp = r >> 1, p = r & 1;
  Bh[(kp * PSTR + c) * 2 + p] = h;
  Bl[(kp * PSTR + c) * 2 + p] = l;
}

// ---------------- kernel: row LayerNorm (one row per 128-thr block) ---------
__global__ void ln_rows_kernel(const float* __restrict__ X,
                               const float* __restrict__ g,
                               const float* __restrict__ b,
                               float* __restrict__ Y) {
  __shared__ float s1[128];
  __shared__ float s2[128];
  const int r = blockIdx.x, t = threadIdx.x;
  const float v = X[r * D_MODEL + t];
  s1[t] = v; s2[t] = v * v;
  __syncthreads();
  for (int off = 64; off > 0; off >>= 1) {
    if (t < off) { s1[t] += s1[t + off]; s2[t] += s2[t + off]; }
    __syncthreads();
  }
  const float mu  = s1[0] * (1.0f / 128.0f);
  const float var = s2[0] * (1.0f / 128.0f) - mu * mu;
  const float rs  = rsqrtf(var + 1e-5f);
  Y[r * D_MODEL + t] = (v - mu) * rs * g[t] + b[t];
}

// ---------------- kernel: [2048,128]x[128,128] GEMM (+bias, opt LN+GELU) ----
// f32 WMMA path (front-end is <1% of FLOPs; keep exact).
__global__ void gemm128_kernel(const float* __restrict__ X,
                               const float* __restrict__ W,
                               const float* __restrict__ bias,
                               const float* __restrict__ g,
                               const float* __restrict__ be,
                               float* __restrict__ Y,
                               int mode) { // 0 = plain, 1 = LN+GELU epilogue
  extern __shared__ float lds[];
  float* Ws = lds;                     // 128 * LDSS
  float* Xs = Ws + 128 * LDSS;         // 16  * LDSS
  float* Ys = Xs + 16 * LDSS;          // 16  * LDSS
  __shared__ float red[2][16][16];

  const int tid  = threadIdx.x;
  const int row0 = blockIdx.x * 16;

  for (int i = tid; i < 128 * 128; i += 256) {
    const int r = i >> 7, c = i & 127;
    Ws[r * LDSS + c] = W[i];
  }
  for (int i = tid; i < 16 * 128; i += 256) {
    const int r = i >> 7, c = i & 127;
    Xs[r * LDSS + c] = X[(row0 + r) * D_MODEL + c];
  }
  __syncthreads();

  {
    const int lane = tid & 31, wid = tid >> 5;
    const int half = lane >> 4, l15 = lane & 15, kb = half * 2;
    const int col  = wid * 16 + l15;         // each wave owns one 16-col tile
    v8f acc = {};
    #pragma unroll 8
    for (int k0 = 0; k0 < D_MODEL; k0 += 4) {
      v2f a, b;
      a.x = Xs[l15 * LDSS + k0 + kb];
      a.y = Xs[l15 * LDSS + k0 + kb + 1];
      b.x = Ws[(k0 + kb) * LDSS + col];
      b.y = Ws[(k0 + kb + 1) * LDSS + col];
      acc = wmma_f32_4(a, b, acc);
    }
    #pragma unroll
    for (int v = 0; v < 8; ++v)
      Ys[(v + half * 8) * LDSS + col] = acc[v];
  }
  __syncthreads();

  const int r = tid >> 4, gx = tid & 15;
  float* yr = Ys + r * LDSS;
  float loc[8];
  float s = 0.f, sq = 0.f;
  #pragma unroll
  for (int j = 0; j < 8; ++j) {
    const int c = gx * 8 + j;
    const float v = yr[c] + bias[c];
    loc[j] = v; s += v; sq += v * v;
  }
  red[0][r][gx] = s; red[1][r][gx] = sq;
  __syncthreads();
  float mu = 0.f, rs = 0.f;
  if (mode == 1) {
    float ts = 0.f, tq = 0.f;
    #pragma unroll
    for (int j = 0; j < 16; ++j) { ts += red[0][r][j]; tq += red[1][r][j]; }
    mu = ts * (1.0f / 128.0f);
    const float var = tq * (1.0f / 128.0f) - mu * mu;
    rs = rsqrtf(var + 1e-5f);
  }
  #pragma unroll
  for (int j = 0; j < 8; ++j) {
    const int c = gx * 8 + j;
    float v = loc[j];
    if (mode == 1) v = gelu_exact((v - mu) * rs * g[c] + be[c]);
    Y[(row0 + r) * D_MODEL + c] = v;
  }
}

// ---------------- kernel: delta = softplus(dnh.w3 + b3); Bx = Bd0*delta*xn --
__global__ void delta_bx_kernel(const float* __restrict__ dnh,
                                const float* __restrict__ dnw3,
                                const float* __restrict__ dnb3,
                                const float* __restrict__ bd0,
                                const float* __restrict__ xn,
                                float* __restrict__ delta,
                                float* __restrict__ bx) {
  __shared__ float s1[128];
  const int r = blockIdx.x, t = threadIdx.x;
  s1[t] = dnh[r * D_MODEL + t] * dnw3[t];
  __syncthreads();
  for (int off = 64; off > 0; off >>= 1) {
    if (t < off) s1[t] += s1[t + off];
    __syncthreads();
  }
  const float z = s1[0] + dnb3[0];
  const float d = (z > 20.f) ? z : log1pf(expf(z));
  if (t == 0) delta[r] = d;
  bx[r * D_MODEL + t] = bd0[r * D_MODEL + t] * d * xn[r * D_MODEL + t];
}

// ---------------- kernel: ||A||_inf (max abs row sum) -----------------------
__global__ void anorm_kernel(const float* __restrict__ A, float* __restrict__ normA) {
  __shared__ float s1[128];
  const int t = threadIdx.x;
  float acc = 0.f;
  for (int c = 0; c < D_MODEL; ++c) acc += fabsf(A[t * D_MODEL + c]);
  s1[t] = acc;
  __syncthreads();
  for (int off = 64; off > 0; off >>= 1) {
    if (t < off) s1[t] = fmaxf(s1[t], s1[t + off]);
    __syncthreads();
  }
  if (t == 0) normA[0] = s1[0];
}

// =====================================================================
//  expm(delta*A): scaling + degree-8 Taylor (Horner) + squaring,
//  all matmuls on the bf16 WMMA pipe with bf16x3 split precision.
//  One 256-thread block per matrix.  LDS ~272 KB:
//    ThB/TlB : T  (B-layout hi/lo)      2 x 64*PSTR dwords
//    Dm      : f32 matmul destination       128*LDSS f32
//    PhA/PlA : P  (A-layout hi/lo)      2 x 128*LDSB bf16
//    PhB/PlB : P  (B-layout hi/lo)      2 x 64*PSTR dwords
//  Horner uses P as A-operand, T as B-operand (T,P commute: P = poly(T)).
// =====================================================================
__global__ void expm_kernel(const float* __restrict__ A,
                            const float* __restrict__ delta,
                            const float* __restrict__ normA,
                            float* __restrict__ adbuf,   // [B*CHUNK][128*128]
                            int chunk0) {
  extern __shared__ char ldsraw[];
  unsigned* ThB = (unsigned*)ldsraw;                 // 64*PSTR dwords
  unsigned* TlB = ThB + 64 * PSTR;                   // 64*PSTR dwords
  float*    Dm  = (float*)(TlB + 64 * PSTR);         // 128*LDSS f32
  __bf16*   PhA = (__bf16*)(Dm + 128 * LDSS);        // 128*LDSB bf16
  __bf16*   PlA = PhA + 128 * LDSB;
  unsigned* PhB = (unsigned*)(PlA + 128 * LDSB);     // 64*PSTR dwords
  unsigned* PlB = PhB + 64 * PSTR;

  const int b   = blockIdx.x / CHUNK;
  const int tt  = blockIdx.x % CHUNK;
  const int row = b * SEQ + chunk0 + tt;
  const int tid = threadIdx.x;

  const float d   = delta[row];
  const float nrm = d * normA[0];
  int s = 0;
  if (nrm > 0.25f) {
    s = (int)ceilf(log2f(nrm * 4.0f));
    if (s < 0) s = 0;
    if (s > 30) s = 30;
  }
  const float scl = d * exp2f(-(float)s);

  // T = (delta/2^s)*A  (B-layout hi/lo) ; P0 = I + T/8 (both layouts hi/lo)
  for (int i = tid; i < 128 * 128; i += 256) {
    const int r = i >> 7, c = i & 127;
    const float tv = A[i] * scl;
    {
      const __bf16 h = (__bf16)tv;
      const __bf16 l = (__bf16)(tv - (float)h);
      const int kp = r >> 1, p = r & 1;
      ((__bf16*)ThB)[(kp * PSTR + c) * 2 + p] = h;
      ((__bf16*)TlB)[(kp * PSTR + c) * 2 + p] = l;
    }
    const float pv = tv * 0.125f + ((r == c) ? 1.0f : 0.0f);
    split_store(pv, r, c, PhA, PlA, (__bf16*)PhB, (__bf16*)PlB);
  }
  __syncthreads();

  // Horner: P <- I + (P*T)/k  for k = 7..1  => degree-8 Taylor of exp(T)
  for (int k = 7; k >= 1; --k) {
    wg_matmul_bf16x3(PhA, PlA, ThB, TlB, Dm, 1.0f / (float)k, /*addI=*/true);
    __syncthreads();
    for (int i = tid; i < 128 * 128; i += 256) {
      const int r = i >> 7, c = i & 127;
      split_store(Dm[r * LDSS + c], r, c, PhA, PlA, (__bf16*)PhB, (__bf16*)PlB);
    }
    __syncthreads();
  }
  // squaring: P <- P*P, s times
  for (int i = 0; i < s; ++i) {
    wg_matmul_bf16x3(PhA, PlA, PhB, PlB, Dm, 1.0f, /*addI=*/false);
    __syncthreads();
    for (int j = tid; j < 128 * 128; j += 256) {
      const int r = j >> 7, c = j & 127;
      split_store(Dm[r * LDSS + c], r, c, PhA, PlA, (__bf16*)PhB, (__bf16*)PlB);
    }
    __syncthreads();
  }

  float* out = adbuf + (size_t)blockIdx.x * (D_MODEL * D_MODEL);
  for (int i = tid; i < 128 * 128; i += 256) {
    const int r = i >> 7, c = i & 127;
    out[i] = Dm[r * LDSS + c];
  }
}

// ---------------- kernel: sequential scan over one chunk (1 block / batch) --
__global__ void scan_kernel(const float* __restrict__ adbuf,
                            const float* __restrict__ bx,
                            float* __restrict__ hstate,
                            float* __restrict__ hs,
                            int chunk0) {
  __shared__ float h[128];
  const int b = blockIdx.x, t = threadIdx.x;
  h[t] = hstate[b * D_MODEL + t];
  __syncthreads();

  for (int step = 0; step < CHUNK; ++step) {
    const float* Prow = adbuf + ((size_t)(b * CHUNK + step)) * (D_MODEL * D_MODEL) + t * D_MODEL;
    if (step + 1 < CHUNK)
      __builtin_prefetch(Prow + D_MODEL * D_MODEL, 0, 1);  // global_prefetch_b8
    float acc = 0.f;
    #pragma unroll 8
    for (int k = 0; k < D_MODEL; k += 4) {
      const float4 p = *(const float4*)(Prow + k);
      acc += p.x * h[k] + p.y * h[k + 1] + p.z * h[k + 2] + p.w * h[k + 3];
    }
    const int row = b * SEQ + chunk0 + step;
    const float nh = h[t] + acc + bx[row * D_MODEL + t]; // h <- h + A_d h + B_d x
    __syncthreads();
    h[t] = nh;
    hs[row * D_MODEL + t] = nh;
    __syncthreads();
  }
  hstate[b * D_MODEL + t] = h[t];
}

__global__ void init_state_kernel(const float* __restrict__ hprev, float* __restrict__ hstate) {
  const int t = blockIdx.x * blockDim.x + threadIdx.x;
  if (t < BATCH * D_MODEL) hstate[t] = hprev[t];
}

__global__ void copy_hfinal_kernel(const float* __restrict__ hstate, float* __restrict__ out) {
  const int t = threadIdx.x;
  if (t < BATCH * D_MODEL) out[t] = hstate[t];
}

// ---------------- host launcher ---------------------------------------------
extern "C" void kernel_launch(void* const* d_in, const int* in_sizes, int n_in,
                              void* d_out, int out_size, void* d_ws, size_t ws_size,
                              hipStream_t stream) {
  const float* x      = (const float*)d_in[0];
  const float* h_prev = (const float*)d_in[1];
  const float* A      = (const float*)d_in[2];
  const float* dn_w1  = (const float*)d_in[3];
  const float* dn_b1  = (const float*)d_in[4];
  const float* dn_g1  = (const float*)d_in[5];
  const float* dn_be1 = (const float*)d_in[6];
  const float* dn_w2  = (const float*)d_in[7];
  const float* dn_b2  = (const float*)d_in[8];
  const float* dn_g2  = (const float*)d_in[9];
  const float* dn_be2 = (const float*)d_in[10];
  const float* bn_w1  = (const float*)d_in[11];
  const float* bn_b1  = (const float*)d_in[12];
  const float* bn_g1  = (const float*)d_in[13];
  const float* bn_be1 = (const float*)d_in[14];
  const float* bn_w2  = (const float*)d_in[15];
  const float* bn_b2  = (const float*)d_in[16];
  const float* bn_g2  = (const float*)d_in[17];
  const float* bn_be2 = (const float*)d_in[18];
  const float* dn_w3  = (const float*)d_in[19];
  const float* dn_b3  = (const float*)d_in[20];
  const float* bn_w3  = (const float*)d_in[21];
  const float* bn_b3  = (const float*)d_in[22];
  const float* n1_g   = (const float*)d_in[23];
  const float* n1_b   = (const float*)d_in[24];
  const float* n2_g   = (const float*)d_in[25];
  const float* n2_b   = (const float*)d_in[26];
  const float* out_w  = (const float*)d_in[27];
  const float* out_b  = (const float*)d_in[28];

  float* y_out = (float*)d_out;                         // [2,1024,128]
  float* hfin  = y_out + ROWS * D_MODEL;                // [2,128]

  // workspace layout (floats)
  float* ws = (float*)d_ws;
  const size_t RD = (size_t)ROWS * D_MODEL;             // 262144
  float* xn     = ws;
  float* t0     = xn + RD;
  float* t1     = t0 + RD;
  float* t2     = t1 + RD;
  float* t3     = t2 + RD;
  float* t4     = t3 + RD;
  float* bx     = t4 + RD;
  float* hs     = bx + RD;
  float* hsn    = hs + RD;
  float* deltaw = hsn + RD;                             // 2048
  float* normA  = deltaw + ROWS;                        // 1
  float* hstate = normA + 8;                            // 256
  float* adbuf  = hstate + 256;                         // 2*CHUNK*128*128 = 32 MB

  const int GEMM_LDS = (128 * LDSS + 2 * 16 * LDSS) * (int)sizeof(float);      // ~84.5 KB
  const int EXPM_LDS = (4 * 64 * PSTR * 4)          /* ThB,TlB,PhB,PlB */
                     + (128 * LDSS * 4)             /* Dm               */
                     + (2 * 128 * LDSB * 2);        /* PhA,PlA          */  // ~272 KB
  hipFuncSetAttribute(reinterpret_cast<const void*>(gemm128_kernel),
                      hipFuncAttributeMaxDynamicSharedMemorySize, GEMM_LDS);
  hipFuncSetAttribute(reinterpret_cast<const void*>(expm_kernel),
                      hipFuncAttributeMaxDynamicSharedMemorySize, EXPM_LDS);

  // 1) xn = LN(x; n1)
  ln_rows_kernel<<<ROWS, 128, 0, stream>>>(x, n1_g, n1_b, xn);

  // 2) delta_net two layers
  gemm128_kernel<<<ROWS / 16, 256, GEMM_LDS, stream>>>(xn, dn_w1, dn_b1, dn_g1, dn_be1, t0, 1);
  gemm128_kernel<<<ROWS / 16, 256, GEMM_LDS, stream>>>(t0, dn_w2, dn_b2, dn_g2, dn_be2, t1, 1);

  // 3) B_net two layers + final projection (plain)
  gemm128_kernel<<<ROWS / 16, 256, GEMM_LDS, stream>>>(xn, bn_w1, bn_b1, bn_g1, bn_be1, t2, 1);
  gemm128_kernel<<<ROWS / 16, 256, GEMM_LDS, stream>>>(t2, bn_w2, bn_b2, bn_g2, bn_be2, t3, 1);
  gemm128_kernel<<<ROWS / 16, 256, GEMM_LDS, stream>>>(t3, bn_w3, bn_b3, bn_g2, bn_be2, t4, 0);

  // 4) delta = softplus(t1 . dn_w3 + dn_b3); Bx = t4 * delta * xn
  delta_bx_kernel<<<ROWS, 128, 0, stream>>>(t1, dn_w3, dn_b3, t4, xn, deltaw, bx);

  // 5) ||A||_inf for scaling
  anorm_kernel<<<1, 128, 0, stream>>>(A, normA);

  // 6) chunked: expm of 512 matrices (bf16x3 WMMA), then sequential scan
  init_state_kernel<<<1, 256, 0, stream>>>(h_prev, hstate);
  for (int c = 0; c < NCHUNK; ++c) {
    expm_kernel<<<BATCH * CHUNK, 256, EXPM_LDS, stream>>>(A, deltaw, normA, adbuf, c * CHUNK);
    scan_kernel<<<BATCH, 128, 0, stream>>>(adbuf, bx, hstate, hs, c * CHUNK);
  }
  copy_hfinal_kernel<<<1, 256, 0, stream>>>(hstate, hfin);

  // 7) y = LN(hs; n2) @ out_w + out_b
  ln_rows_kernel<<<ROWS, 128, 0, stream>>>(hs, n2_g, n2_b, hsn);
  gemm128_kernel<<<ROWS / 16, 256, GEMM_LDS, stream>>>(hsn, out_w, out_b, n2_g, n2_b, y_out, 0);

  (void)in_sizes; (void)n_in; (void)out_size; (void)ws_size;
}
